// ChunkedTTTLayer_54022098649264
// MI455X (gfx1250) — matrix operand order repaired and checked by reading, not compiled, not run on hardware
//
#include <hip/hip_runtime.h>
#include <hip/hip_bf16.h>

#define HIDDEN_ 768
#define TTTH_   2048
#define CHUNK_  512
#define SEQ_    8192
#define BATCH_  4
#define MROWS_  2048          // BATCH_*CHUNK_ (rows per chunk, batch-flattened)
#define NCHUNK_ 16
#define LR_     0.01f

typedef __attribute__((ext_vector_type(16))) __bf16 v16bf;
typedef __attribute__((ext_vector_type(8)))  float  v8f;

// Row mapping for chunked x: flattened row m in [0,2048) -> batch m/512, seq c*512 + m%512
__device__ __forceinline__ long xrow(int m, long cb) {
    return cb + (long)(m >> 9) * ((long)SEQ_ * HIDDEN_) + (long)(m & 511) * HIDDEN_;
}

// Scatter one f32 element into the ISA A-fragment layout (16-bit A 16x32, §7.12.2)
__device__ __forceinline__ void putA(__bf16 (*AsF)[32][16], int m, int k, float v) {
    int half = (k >> 3) & 1;
    int lane = (m & 15) | (half << 4);
    int e    = ((k >> 4) << 3) | (k & 7);
    AsF[m >> 4][lane][e] = (__bf16)v;
}
// B-fragment layout (16-bit B 32x16: lanes 0-15 K=0..7,16..23, lanes 16-31 K=8..15,24..31)
__device__ __forceinline__ void putB(__bf16 (*BsF)[32][16], int k, int n, float v) {
    int lane = (n & 15) | (((k >> 3) & 1) << 4);
    BsF[n >> 4][lane][((k >> 4) << 3) | (k & 7)] = (__bf16)v;
}

// AM: 0 = A[m*lda+k], 1 = A[k*lda+m], 2 = x chunked rows, 3 = x chunked transposed
// BM: 0 = B[k*ldb+n], 1 = B[n*ldb+k]
// EPI: 0 store, 1 store G + A=silu(H0)*G, 2 out + dOut + loss, 3 dSwiGLU in place, 4 SGD update
template<int AM, int BM, int EPI>
__global__ __launch_bounds__(256) void ttt_gemm(
    const float* __restrict__ A, long lda,
    const float* __restrict__ B, long ldb,
    float* __restrict__ C, long ldc,
    int M, int N, int K,
    long xcb, const float* __restrict__ Xf,
    float* __restrict__ P0, float* __restrict__ P1,
    float* __restrict__ lossSlot, float dscale)
{
    (void)M;
    const int tid   = threadIdx.x;
    const int lane  = tid & 31;
    const int wid   = tid >> 5;     // 8 waves
    const int waveM = wid >> 1;     // 0..3 -> 32-row strips
    const int waveN = wid & 1;      // 0..1 -> 64-col strips
    const int bm = blockIdx.y * 128;
    const int bn = blockIdx.x * 128;

    __shared__ alignas(32) __bf16 AsF[8][32][16];   // 8 M-subtiles, fragment order
    __shared__ alignas(32) __bf16 BsF[8][32][16];   // 8 N-subtiles, fragment order
    __shared__ float redbuf[256];

    v8f acc[2][4];
#pragma unroll
    for (int i = 0; i < 2; ++i)
#pragma unroll
        for (int j = 0; j < 4; ++j)
            acc[i][j] = (v8f){0.f,0.f,0.f,0.f,0.f,0.f,0.f,0.f};

    for (int k0 = 0; k0 < K; k0 += 32) {
        __syncthreads();   // previous tile fully consumed

        // ---- stage A tile (128 x 32) into fragment-ordered LDS, fp32 -> bf16 ----
        if (AM == 0 || AM == 2) {              // k contiguous in memory
            int m  = tid >> 1;
            int kk = (tid & 1) << 4;
            long rowoff = (AM == 0) ? (long)(bm + m) * lda : xrow(bm + m, xcb);
            const float4* p = reinterpret_cast<const float4*>(A + rowoff + k0 + kk);
#pragma unroll
            for (int q = 0; q < 4; ++q) {
                float4 v = p[q];
                putA(AsF, m, kk + q*4 + 0, v.x);
                putA(AsF, m, kk + q*4 + 1, v.y);
                putA(AsF, m, kk + q*4 + 2, v.z);
                putA(AsF, m, kk + q*4 + 3, v.w);
            }
            if (k0 + 32 < K) __builtin_prefetch(p + 8, 0, 1);   // next K-tile, same row
        } else {                                // m contiguous in memory (transposed A)
            int kk = tid >> 3;
            int m0 = (tid & 7) << 4;
            long rowoff = (AM == 1) ? (long)(k0 + kk) * lda : xrow(k0 + kk, xcb);
            const float4* p = reinterpret_cast<const float4*>(A + rowoff + bm + m0);
#pragma unroll
            for (int q = 0; q < 4; ++q) {
                float4 v = p[q];
                putA(AsF, m0 + q*4 + 0, kk, v.x);
                putA(AsF, m0 + q*4 + 1, kk, v.y);
                putA(AsF, m0 + q*4 + 2, kk, v.z);
                putA(AsF, m0 + q*4 + 3, kk, v.w);
            }
        }

        // ---- stage B tile (32 x 128) ----
        if (BM == 0) {                          // n contiguous
            int kk = tid >> 3;
            int n0 = (tid & 7) << 4;
            const float4* p = reinterpret_cast<const float4*>(B + (long)(k0 + kk) * ldb + bn + n0);
#pragma unroll
            for (int q = 0; q < 4; ++q) {
                float4 v = p[q];
                putB(BsF, kk, n0 + q*4 + 0, v.x);
                putB(BsF, kk, n0 + q*4 + 1, v.y);
                putB(BsF, kk, n0 + q*4 + 2, v.z);
                putB(BsF, kk, n0 + q*4 + 3, v.w);
            }
            if (k0 + 32 < K) __builtin_prefetch(reinterpret_cast<const float*>(p) + 32 * ldb, 0, 1);
        } else {                                // k contiguous (B transposed)
            int n   = tid >> 1;
            int kk0 = (tid & 1) << 4;
            const float4* p = reinterpret_cast<const float4*>(B + (long)(bn + n) * ldb + k0 + kk0);
#pragma unroll
            for (int q = 0; q < 4; ++q) {
                float4 v = p[q];
                putB(BsF, kk0 + q*4 + 0, n, v.x);
                putB(BsF, kk0 + q*4 + 1, n, v.y);
                putB(BsF, kk0 + q*4 + 2, n, v.z);
                putB(BsF, kk0 + q*4 + 3, n, v.w);
            }
            if (k0 + 32 < K) __builtin_prefetch(p + 8, 0, 1);
        }

        __syncthreads();

        // ---- WMMA: each wave does 2x4 tiles of 16x16, one K=32 step ----
        v16bf afrag0 = *reinterpret_cast<const v16bf*>(&AsF[waveM * 2 + 0][lane][0]);
        v16bf afrag1 = *reinterpret_cast<const v16bf*>(&AsF[waveM * 2 + 1][lane][0]);
#pragma unroll
        for (int j = 0; j < 4; ++j) {
            v16bf bfrag = *reinterpret_cast<const v16bf*>(&BsF[waveN * 4 + j][lane][0]);
            acc[0][j] = __builtin_amdgcn_wmma_f32_16x16x32_bf16(
                false, afrag0, false, bfrag, (short)0, acc[0][j], false, false);
            acc[1][j] = __builtin_amdgcn_wmma_f32_16x16x32_bf16(
                false, afrag1, false, bfrag, (short)0, acc[1][j], false, false);
        }
    }

    // ---- fused epilogues (C/D layout: VGPR r, lane -> M=(lane/16)*8+r, N=lane%16) ----
    float lsum = 0.f;
#pragma unroll
    for (int i = 0; i < 2; ++i) {
#pragma unroll
        for (int j = 0; j < 4; ++j) {
            int rbase = bm + waveM * 32 + i * 16 + ((lane >> 4) << 3);
            int col   = bn + waveN * 64 + j * 16 + (lane & 15);
#pragma unroll
            for (int r = 0; r < 8; ++r) {
                int   row = rbase + r;
                float d   = acc[i][j][r];
                if constexpr (EPI == 0) {            // plain store (H0)
                    C[(long)row * ldc + col] = d;
                } else if constexpr (EPI == 1) {     // store G and A = silu(H0)*G
                    long idx = (long)row * ldc + col;
                    C[idx] = d;
                    float h   = P0[idx];
                    float sig = 1.f / (1.f + __expf(-h));
                    P1[idx] = h * sig * d;
                } else if constexpr (EPI == 2) {     // Out + dOut + loss partial
                    long oi = xrow(row, xcb) + col;
                    C[oi] = d;
                    float diff = d - Xf[oi];
                    lsum += diff * diff;
                    P0[(long)row * N + col] = dscale * diff;
                } else if constexpr (EPI == 3) {     // D = dA; overwrite H0->dH0, G->dG
                    long idx = (long)row * ldc + col;
                    float h = P0[idx], g = P1[idx];
                    float sig  = 1.f / (1.f + __expf(-h));
                    P0[idx] = d * g * sig * (1.f + h * (1.f - sig));
                    P1[idx] = d * (h * sig);
                } else {                             // EPI == 4: fused SGD
                    long idx = (long)row * ldc + col;
                    C[idx] -= LR_ * d;
                }
            }
        }
    }

    if constexpr (EPI == 2) {                    // deterministic block reduction
        redbuf[tid] = lsum;
        __syncthreads();
#pragma unroll
        for (int s = 128; s > 0; s >>= 1) {
            if (tid < s) redbuf[tid] += redbuf[tid + s];
            __syncthreads();
        }
        if (tid == 0) lossSlot[blockIdx.y * gridDim.x + blockIdx.x] = redbuf[0];
    }
}

__global__ void ttt_finalize(const float* __restrict__ lossP, float* __restrict__ outLoss) {
    float s = 0.f;
    for (int i = 0; i < NCHUNK_ * 96; ++i) s += lossP[i];   // fixed order -> deterministic
    *outLoss = s / (float)NCHUNK_ / (float)((long)MROWS_ * HIDDEN_);
}

extern "C" void kernel_launch(void* const* d_in, const int* in_sizes, int n_in,
                              void* d_out, int out_size, void* d_ws, size_t ws_size,
                              hipStream_t stream) {
    (void)in_sizes; (void)n_in; (void)out_size; (void)ws_size;
    const float* x  = (const float*)d_in[0];
    float* out = (float*)d_out;
    float* ws  = (float*)d_ws;

    const long WSZ = (long)HIDDEN_ * TTTH_;      // 1572864
    const long HSZ = (long)MROWS_ * TTTH_;       // 4194304
    const long OSZ = (long)MROWS_ * HIDDEN_;     // 1572864
    float* w0c = ws;
    float* w1c = w0c + WSZ;
    float* w2c = w1c + WSZ;
    float* H0  = w2c + WSZ;
    float* G   = H0 + HSZ;
    float* Ab  = G + HSZ;
    float* dO  = Ab + HSZ;
    float* lossP = dO + OSZ;                     // NCHUNK_*96 slots

    hipMemcpyAsync(w0c, d_in[1], WSZ * sizeof(float), hipMemcpyDeviceToDevice, stream);
    hipMemcpyAsync(w1c, d_in[2], WSZ * sizeof(float), hipMemcpyDeviceToDevice, stream);
    hipMemcpyAsync(w2c, d_in[3], WSZ * sizeof(float), hipMemcpyDeviceToDevice, stream);

    const dim3 blk(256);
    const float dscale = 2.f / (float)((long)MROWS_ * HIDDEN_);

    for (int c = 0; c < NCHUNK_; ++c) {
        long cb = (long)c * CHUNK_ * HIDDEN_;
        // 1) H0 = Xc @ w0
        ttt_gemm<2,0,0><<<dim3(16,16), blk, 0, stream>>>(x, 0, w0c, TTTH_, H0, TTTH_,
            MROWS_, TTTH_, HIDDEN_, cb, x, nullptr, nullptr, nullptr, 0.f);
        // 2) G = Xc @ w2 ; A = silu(H0)*G
        ttt_gemm<2,0,1><<<dim3(16,16), blk, 0, stream>>>(x, 0, w2c, TTTH_, G, TTTH_,
            MROWS_, TTTH_, HIDDEN_, cb, x, H0, Ab, nullptr, 0.f);
        // 3) Out = A @ w1 ; dOut ; loss partials
        ttt_gemm<0,0,2><<<dim3(6,16), blk, 0, stream>>>(Ab, TTTH_, w1c, HIDDEN_, out, 0,
            MROWS_, HIDDEN_, TTTH_, cb, x, dO, nullptr, lossP + c * 96, dscale);
        // 4) dA = dOut @ w1^T ; H0<-dH0, G<-dG (fused SwiGLU backward)
        ttt_gemm<0,1,3><<<dim3(16,16), blk, 0, stream>>>(dO, HIDDEN_, w1c, HIDDEN_, nullptr, TTTH_,
            MROWS_, TTTH_, HIDDEN_, cb, x, H0, G, nullptr, 0.f);
        // 5) w1 -= LR * (A^T @ dOut)
        ttt_gemm<1,0,4><<<dim3(6,16), blk, 0, stream>>>(Ab, TTTH_, dO, HIDDEN_, w1c, HIDDEN_,
            TTTH_, HIDDEN_, MROWS_, cb, x, nullptr, nullptr, nullptr, 0.f);
        // 6) w0 -= LR * (Xc^T @ dH0)
        ttt_gemm<3,0,4><<<dim3(16,6), blk, 0, stream>>>(x, 0, H0, TTTH_, w0c, TTTH_,
            HIDDEN_, TTTH_, MROWS_, cb, x, nullptr, nullptr, nullptr, 0.f);
        // 7) w2 -= LR * (Xc^T @ dG)
        ttt_gemm<3,0,4><<<dim3(16,6), blk, 0, stream>>>(x, 0, G, TTTH_, w2c, TTTH_,
            HIDDEN_, TTTH_, MROWS_, cb, x, nullptr, nullptr, nullptr, 0.f);
    }

    ttt_finalize<<<1, 1, 0, stream>>>(lossP, out + (long)BATCH_ * SEQ_ * HIDDEN_);
}
